// Model_1_10995116278167
// MI455X (gfx1250) — compile-verified, matrix operand order checked by reading, not run on hardware
//
#include <hip/hip_runtime.h>
#include <hip/hip_bf16.h>
#include <math.h>

// ---------------- problem constants (match reference) ----------------
#define NNODES  65536
#define NEDGES  524288
#define NMSG    (NEDGES + NNODES)   // edges + self loops = 589824
#define FIN     64
#define FH      128
#define NB      128                 // graphs
#define NPG     512                 // nodes per graph
#define NKEEP   256                 // topk kept per graph

typedef __attribute__((ext_vector_type(16))) _Float16 v16h;
typedef __attribute__((ext_vector_type(8)))  float    v8f;

__device__ __forceinline__ float lrelu(float v, float s) { return v > 0.f ? v : v * s; }

// float atomic-max via int/uint monotone punning (m initialized to -inf)
__device__ __forceinline__ void atomicMaxF(float* addr, float v) {
    if (v >= 0.f) atomicMax((int*)addr, __float_as_int(v));
    else          atomicMin((unsigned int*)addr, __float_as_uint(v));
}

// =====================================================================
// WMMA GEMM: out[M,N] = leaky(A[M,K] @ W[K,N] + bias[N], slope)
// N == NTILES*16.  One wave computes a 16 x N strip: A fragment loaded
// once per K-chunk, W staged in LDS pre-swizzled into B-fragment layout
// (each lane's 16 halfs contiguous -> aligned 32B ds reads).
// A,W fp32 in memory, f16 fragments, f32 accumulate. M%16==0, K%32==0.
// =====================================================================
template<int NTILES>
__global__ void gemm_wmma(const float* __restrict__ A, const float* __restrict__ W,
                          const float* __restrict__ bias, float* __restrict__ out,
                          int M, int K, float slope)
{
    extern __shared__ v16h wfrag[];              // [K/32][NTILES][32 lanes]
    const int N = NTILES * 16;
    const int tid  = threadIdx.x;
    const int lane = tid & 31;
    const int wave = tid >> 5;
    const int kchunks = K >> 5;

    // -------- cooperative swizzle of W into B-fragment layout --------
    _Float16* wlds = (_Float16*)wfrag;
    const int total = kchunks * NTILES * 512;    // == K*N elements
    for (int e = tid; e < total; e += blockDim.x) {
        int j  = e & 15;                         // half slot within lane
        int ln = (e >> 4) & 31;                  // lane
        int nt = (e >> 9) % NTILES;              // n-tile
        int kc = e / (NTILES * 512);             // k-chunk
        int k  = kc * 32 + (ln >> 4) * 8 + (j & 7) + ((j >> 3) << 4);
        int n  = nt * 16 + (ln & 15);
        wlds[e] = (_Float16)W[(size_t)k * N + n];
    }
    __syncthreads();

    const int tile = blockIdx.x * (blockDim.x >> 5) + wave;   // 16-row M tile
    if (tile >= (M >> 4)) return;

    const int sel  = lane >> 4;                  // K-phase select (ISA f16 A layout)
    const int mrow = (tile << 4) + (lane & 15);

    v8f acc[NTILES];
    #pragma unroll
    for (int nt = 0; nt < NTILES; ++nt) acc[nt] = (v8f){};

    for (int kc = 0; kc < kchunks; ++kc) {
        const float* ap = A + (size_t)mrow * K + kc * 32 + sel * 8;
        __builtin_prefetch(ap + 32, 0, 0);       // global_prefetch_b8 (speculative)
        float4 f0 = *(const float4*)(ap);
        float4 f1 = *(const float4*)(ap + 4);
        float4 f2 = *(const float4*)(ap + 16);
        float4 f3 = *(const float4*)(ap + 20);
        v16h a;
        a[0]=(_Float16)f0.x;  a[1]=(_Float16)f0.y;  a[2]=(_Float16)f0.z;  a[3]=(_Float16)f0.w;
        a[4]=(_Float16)f1.x;  a[5]=(_Float16)f1.y;  a[6]=(_Float16)f1.z;  a[7]=(_Float16)f1.w;
        a[8]=(_Float16)f2.x;  a[9]=(_Float16)f2.y;  a[10]=(_Float16)f2.z; a[11]=(_Float16)f2.w;
        a[12]=(_Float16)f3.x; a[13]=(_Float16)f3.y; a[14]=(_Float16)f3.z; a[15]=(_Float16)f3.w;

        #pragma unroll
        for (int nt = 0; nt < NTILES; ++nt) {
            v16h b = wfrag[(kc * NTILES + nt) * 32 + lane];
            acc[nt] = __builtin_amdgcn_wmma_f32_16x16x32_f16(false, a, false, b,
                                                             (short)0, acc[nt], false, false);
        }
    }

    const int mbase = (tile << 4) + sel * 8;     // C/D layout: rows i + (lane>=16)*8
    #pragma unroll
    for (int nt = 0; nt < NTILES; ++nt) {
        const int n = nt * 16 + (lane & 15);
        const float bv = bias ? bias[n] : 0.f;
        #pragma unroll
        for (int i = 0; i < 8; ++i) {
            float v = acc[nt][i] + bv;
            out[(size_t)(mbase + i) * N + n] = lrelu(v, slope);
        }
    }
}

// =====================================================================
// GATv2 edge phase (3-pass scatter softmax over destinations)
// =====================================================================
__global__ void gat_init(float* __restrict__ h, float* __restrict__ m,
                         float* __restrict__ s, const float* __restrict__ bo)
{
    size_t idx = (size_t)blockIdx.x * blockDim.x + threadIdx.x;
    if (idx < (size_t)NNODES * FH) h[idx] = bo[idx & (FH - 1)];
    if (idx < NNODES) { m[idx] = -INFINITY; s[idx] = 0.f; }
}

__global__ void edge_score_max(const int* __restrict__ src, const int* __restrict__ dst,
                               const float* __restrict__ xl, const float* __restrict__ xr,
                               const float* __restrict__ att,
                               float* __restrict__ e, float* __restrict__ m)
{
    int idx = blockIdx.x * blockDim.x + threadIdx.x;
    if (idx >= NMSG) return;
    int j = idx < NEDGES ? src[idx] : (idx - NEDGES);
    int i = idx < NEDGES ? dst[idx] : (idx - NEDGES);
    const float4* lj = (const float4*)(xl + (size_t)j * FH);
    const float4* ri = (const float4*)(xr + (size_t)i * FH);
    const float4* av = (const float4*)att;
    float acc = 0.f;
    #pragma unroll 8
    for (int q = 0; q < FH / 4; ++q) {
        float4 l = lj[q], r = ri[q], a = av[q];
        float x0 = l.x + r.x, x1 = l.y + r.y, x2 = l.z + r.z, x3 = l.w + r.w;
        x0 = lrelu(x0, 0.2f); x1 = lrelu(x1, 0.2f);
        x2 = lrelu(x2, 0.2f); x3 = lrelu(x3, 0.2f);
        acc += x0 * a.x + x1 * a.y + x2 * a.z + x3 * a.w;
    }
    e[idx] = acc;
    atomicMaxF(m + i, acc);
}

__global__ void edge_expsum(const int* __restrict__ dst, float* __restrict__ e,
                            const float* __restrict__ m, float* __restrict__ s)
{
    int idx = blockIdx.x * blockDim.x + threadIdx.x;
    if (idx >= NMSG) return;
    int i = idx < NEDGES ? dst[idx] : (idx - NEDGES);
    float ex = __expf(e[idx] - m[i]);
    e[idx] = ex;                                  // e now holds exp(e - max)
    atomicAdd(s + i, ex);
}

// one wave32 per message: h[i,:] += alpha * xl[j,:]  (4 dims/lane)
__global__ void edge_aggregate(const int* __restrict__ src, const int* __restrict__ dst,
                               const float* __restrict__ e, const float* __restrict__ s,
                               const float* __restrict__ xl, float* __restrict__ h)
{
    int gid  = blockIdx.x * blockDim.x + threadIdx.x;
    int msg  = gid >> 5;
    int lane = gid & 31;
    if (msg >= NMSG) return;
    int j = msg < NEDGES ? src[msg] : (msg - NEDGES);   // broadcast-coalesced
    int i = msg < NEDGES ? dst[msg] : (msg - NEDGES);
    float alpha = e[msg] / s[i];
    float4 xv = ((const float4*)(xl + (size_t)j * FH))[lane];
    float* hp = h + (size_t)i * FH + lane * 4;
    atomicAdd(hp + 0, alpha * xv.x);
    atomicAdd(hp + 1, alpha * xv.y);
    atomicAdd(hp + 2, alpha * xv.z);
    atomicAdd(hp + 3, alpha * xv.w);
}

// =====================================================================
// post-GAT: h = leaky(h, .01) in place; y = tanh(h . p / ||p||)
// =====================================================================
__global__ void p_normalize(const float* __restrict__ p, float* __restrict__ pn)
{
    __shared__ float red[FH];
    int t = threadIdx.x;                            // 128 threads
    float v = p[t];
    red[t] = v * v; __syncthreads();
    for (int o = 64; o > 0; o >>= 1) { if (t < o) red[t] += red[t + o]; __syncthreads(); }
    pn[t] = p[t] * rsqrtf(red[0]);
}

__global__ void node_post(float* __restrict__ h, const float* __restrict__ pn,
                          float* __restrict__ y)
{
    int gid  = blockIdx.x * blockDim.x + threadIdx.x;
    int node = gid >> 5;                            // one wave32 per node
    int lane = gid & 31;
    if (node >= NNODES) return;
    float4* hp = (float4*)(h + (size_t)node * FH);
    const float4* pp = (const float4*)pn;
    float4 v = hp[lane];                            // 32 lanes x 4 = 128 dims
    v.x = lrelu(v.x, 0.01f); v.y = lrelu(v.y, 0.01f);
    v.z = lrelu(v.z, 0.01f); v.w = lrelu(v.w, 0.01f);
    hp[lane] = v;
    float4 q = pp[lane];
    float dot = v.x * q.x + v.y * q.y + v.z * q.z + v.w * q.w;
    #pragma unroll
    for (int o = 16; o > 0; o >>= 1) dot += __shfl_down(dot, o, 32);
    if (lane == 0) y[node] = tanhf(dot);
}

// =====================================================================
// TopK (LDS bitonic sort of 512) + gated max/mean readout -> z[b, zoff..]
// =====================================================================
__global__ void topk_readout(const float* __restrict__ y, const float* __restrict__ h,
                             float* __restrict__ z, int zoff)
{
    __shared__ float sv[NPG];
    __shared__ int   si[NPG];
    __shared__ float redM[256], redS[256];
    int b = blockIdx.x, t = threadIdx.x;            // 256 threads
    sv[t]       = y[b * NPG + t];       si[t]       = t;
    sv[t + 256] = y[b * NPG + t + 256]; si[t + 256] = t + 256;
    __syncthreads();
    for (int k = 2; k <= NPG; k <<= 1) {
        for (int j = k >> 1; j > 0; j >>= 1) {
            int i   = ((t / j) * (j << 1)) + (t % j);
            int ixj = i + j;
            bool up = ((i & k) == 0);               // overall descending
            float a = sv[i], c = sv[ixj];
            bool sw = up ? (a < c) : (a > c);
            if (sw) { sv[i] = c; sv[ixj] = a; int ti = si[i]; si[i] = si[ixj]; si[ixj] = ti; }
            __syncthreads();
        }
    }
    // readout over kept rows 0..255 (order invariant for max/mean)
    int c = t & 127, half = t >> 7;
    float mx = -INFINITY, sm = 0.f;
    for (int r = half * 128; r < half * 128 + 128; ++r) {
        float f = h[((size_t)b * NPG + si[r]) * FH + c] * sv[r];
        mx = fmaxf(mx, f); sm += f;
    }
    redM[t] = mx; redS[t] = sm; __syncthreads();
    if (t < 128) {
        z[(size_t)b * 512 + zoff + t]       = fmaxf(redM[t], redM[t + 128]);
        z[(size_t)b * 512 + zoff + 128 + t] = (redS[t] + redS[t + 128]) * (1.f / NKEEP);
    }
}

// =====================================================================
// final layer 64 -> 2 with log_softmax
// =====================================================================
__global__ void mlp_final(const float* __restrict__ a2, const float* __restrict__ W3,
                          const float* __restrict__ b3, float* __restrict__ out)
{
    int r = threadIdx.x;                            // 128 threads, 1 block
    if (r >= NB) return;
    const float* row = a2 + r * 64;
    float l0 = b3[0], l1 = b3[1];
    #pragma unroll 8
    for (int k = 0; k < 64; ++k) { float v = row[k]; l0 += v * W3[2 * k]; l1 += v * W3[2 * k + 1]; }
    float mv = fmaxf(l0, l1);
    float ls = mv + logf(__expf(l0 - mv) + __expf(l1 - mv));
    out[2 * r] = l0 - ls; out[2 * r + 1] = l1 - ls;
}

// =====================================================================
// host-side orchestration
// =====================================================================
static void run_branch(const float* x, const int* ei,
                       const float* Wl, const float* bl,
                       const float* Wr, const float* br_,
                       const float* att, const float* bo, const float* p,
                       float* xl, float* xr, float* h, float* e, float* m,
                       float* s, float* y, float* pn, float* z, int zoff,
                       hipStream_t st)
{
    const int* src = ei;
    const int* dst = ei + NEDGES;
    const int mt    = NNODES / 16;                  // 4096 M-tiles
    const int blks  = mt / 8;                       // 8 waves/block
    const size_t lds = (size_t)FIN * FH * 2;        // 16 KB f16 W tile
    gemm_wmma<8><<<blks, 256, lds, st>>>(x, Wl, bl, xl, NNODES, FIN, 1.0f);
    gemm_wmma<8><<<blks, 256, lds, st>>>(x, Wr, br_, xr, NNODES, FIN, 1.0f);
    gat_init<<<((size_t)NNODES * FH + 255) / 256, 256, 0, st>>>(h, m, s, bo);
    edge_score_max<<<(NMSG + 255) / 256, 256, 0, st>>>(src, dst, xl, xr, att, e, m);
    edge_expsum<<<(NMSG + 255) / 256, 256, 0, st>>>(dst, e, m, s);
    edge_aggregate<<<((size_t)NMSG * 32 + 255) / 256, 256, 0, st>>>(src, dst, e, s, xl, h);
    p_normalize<<<1, FH, 0, st>>>(p, pn);
    node_post<<<((size_t)NNODES * 32 + 255) / 256, 256, 0, st>>>(h, pn, y);
    topk_readout<<<NB, 256, 0, st>>>(y, h, z, zoff);
}

extern "C" void kernel_launch(void* const* d_in, const int* in_sizes, int n_in,
                              void* d_out, int out_size, void* d_ws, size_t ws_size,
                              hipStream_t stream)
{
    (void)in_sizes; (void)n_in; (void)out_size; (void)ws_size;
    // inputs in setup_inputs() dict order
    const float* source_x = (const float*)d_in[0];
    const float* target_x = (const float*)d_in[1];
    const int*   src_ei   = (const int*)d_in[2];
    const int*   tgt_ei   = (const int*)d_in[3];
    // d_in[4], d_in[5]: batch vectors (graphs contiguous & equal-sized -> unused)
    const float* Wl_s = (const float*)d_in[6];  const float* bl_s = (const float*)d_in[7];
    const float* Wr_s = (const float*)d_in[8];  const float* br_s = (const float*)d_in[9];
    const float* att_s = (const float*)d_in[10]; const float* bo_s = (const float*)d_in[11];
    const float* p_s   = (const float*)d_in[12];
    const float* Wl_t = (const float*)d_in[13]; const float* bl_t = (const float*)d_in[14];
    const float* Wr_t = (const float*)d_in[15]; const float* br_t = (const float*)d_in[16];
    const float* att_t = (const float*)d_in[17]; const float* bo_t = (const float*)d_in[18];
    const float* p_t   = (const float*)d_in[19];
    const float* W1 = (const float*)d_in[20]; const float* b1 = (const float*)d_in[21];
    const float* W2 = (const float*)d_in[22]; const float* b2 = (const float*)d_in[23];
    const float* W3 = (const float*)d_in[24]; const float* b3 = (const float*)d_in[25];

    // workspace carve-up (floats); buffers reused across the two branches
    float* xl  = (float*)d_ws;                       // NNODES*FH
    float* xr  = xl  + (size_t)NNODES * FH;          // NNODES*FH
    float* h   = xr  + (size_t)NNODES * FH;          // NNODES*FH
    float* e   = h   + (size_t)NNODES * FH;          // NMSG
    float* m   = e   + (size_t)NMSG;                 // NNODES
    float* s   = m   + (size_t)NNODES;               // NNODES
    float* y   = s   + (size_t)NNODES;               // NNODES
    float* pn  = y   + (size_t)NNODES;               // 128 (padded)
    float* z   = pn  + 128;                          // NB*512
    float* a1  = z   + (size_t)NB * 512;             // NB*128
    float* a2  = a1  + (size_t)NB * FH;              // NB*64

    // branch 1: source -> z[:, 0:256]
    run_branch(source_x, src_ei, Wl_s, bl_s, Wr_s, br_s, att_s, bo_s, p_s,
               xl, xr, h, e, m, s, y, pn, z, 0, stream);
    // branch 2: target -> z[:, 256:512]
    run_branch(target_x, tgt_ei, Wl_t, bl_t, Wr_t, br_t, att_t, bo_t, p_t,
               xl, xr, h, e, m, s, y, pn, z, 256, stream);

    // MLP head (WMMA GEMMs) + log_softmax
    gemm_wmma<8><<<1, 256, (size_t)512 * 128 * 2, stream>>>(z,  W1, b1, a1, NB, 512, 0.01f);
    gemm_wmma<4><<<1, 256, (size_t)128 * 64 * 2, stream>>>(a1, W2, b2, a2, NB, 128, 0.01f);
    mlp_final<<<1, 128, 0, stream>>>(a2, W3, b3, (float*)d_out);
}
